// SimpleBiLSTM_83537113907559
// MI455X (gfx1250) — matrix operand order repaired and checked by reading, not compiled, not run on hardware
//
#include <hip/hip_runtime.h>
#include <hip/hip_bf16.h>

// ---------------------------------------------------------------------------
// BiLSTM (2-layer, fwd+bwd) for MI455X / gfx1250, wave32 + WMMA bf16.
//   VOCAB=50000, EMBED=256, HIDDEN=512, BATCH=64, SEQ=512
// Per-timestep kernels:  z = bias + A1@W + h_prev@U  via v_wmma_f32_16x16x32_bf16.
// Work split per 16x16 output tile: 8 waves = 4 gates x 2 K-halves; partial
// f32 tiles meet in LDS; 256 threads finalize one element each (gates + c/h).
// ---------------------------------------------------------------------------

#define VOCAB  50000
#define EMBED  256
#define HID    512
#define NB     64      // batch
#define SEQT   512
#define GATES4 2048    // 4*HID

typedef __bf16 bf16_t;
typedef bf16_t v16bf __attribute__((ext_vector_type(16)));
typedef float  v8f   __attribute__((ext_vector_type(8)));
typedef unsigned int u32x4 __attribute__((ext_vector_type(4)));

union Frag { v16bf v; u32x4 q[2]; };

// ---- scalar helpers --------------------------------------------------------
__device__ __forceinline__ unsigned short f2bf(float f) {
  unsigned int u = __float_as_uint(f);
  u = (u + 0x7FFFu + ((u >> 16) & 1u)) >> 16;   // round-to-nearest-even
  return (unsigned short)u;
}
__device__ __forceinline__ float bf2f(unsigned short b) {
  return __uint_as_float(((unsigned int)b) << 16);
}
__device__ __forceinline__ float sigm(float x) {
  return 1.0f / (1.0f + __expf(-x));
}
__device__ __forceinline__ float tanh1(float x) {
  float ax = fabsf(x);
  float e  = __expf(-2.0f * ax);
  float r  = (1.0f - e) / (1.0f + e);
  return copysignf(r, x);
}

// ---- WMMA partial GEMM -----------------------------------------------------
// Accumulate chunks [kb0,kb1) of A(16x32-per-chunk) @ B(32x16 frag-major) into
// one gate's 16x16 f32 tile.  A rows are row-major (per-lane row = lane&15,
// base folded into arow).  B chunks are contiguous 1KB fragments.
__device__ __forceinline__ void gemm_part(v8f& acc,
                                          const unsigned short* __restrict__ arow,
                                          const unsigned short* __restrict__ Bp,
                                          int kbN, int kb0, int kb1,
                                          int nbg, int lane, int koffA) {
  for (int kb = kb0; kb < kb1; ++kb) {
    Frag a;
    const unsigned short* ap = arow + kb * 32 + koffA;
    a.q[0] = *reinterpret_cast<const u32x4*>(ap);
    a.q[1] = *reinterpret_cast<const u32x4*>(ap + 16);
    const unsigned short* bp = Bp + ((size_t)nbg * kbN + kb) * 512 + lane * 16;
    __builtin_prefetch(bp + 512, 0, 1);        // next K-chunk -> global_prefetch_b8
    Frag b;
    b.q[0] = *reinterpret_cast<const u32x4*>(bp);
    b.q[1] = *reinterpret_cast<const u32x4*>(bp + 8);
    acc = __builtin_amdgcn_wmma_f32_16x16x32_bf16(
        false, a.v, false, b.v, (short)0, acc, false, false);
  }
}

// ---- shared epilogue: stash partial tiles, combine gates, update state -----
// zbuf[wave][lane][r]: wave = s*4+g holds gate g, K-half s.
// C/D layout: VGPR r, lane l -> row M = r + 8*(l>=16), col N = l&15.
__device__ __forceinline__ void tile_epilogue(float (*zbuf)[32][8], v8f acc,
                                              int wave, int lane,
                                              float* __restrict__ c,
                                              unsigned short* __restrict__ hout,
                                              int mb, int nb, int tidx) {
#pragma unroll
  for (int r = 0; r < 8; ++r) zbuf[wave][lane][r] = acc[r];
  __syncthreads();
  // one element per thread: tidx in [0,256) -> (row, col) of the 16x16 tile
  int row = tidx >> 4, col = tidx & 15;
  int ls  = col + ((row & 8) ? 16 : 0);
  int r   = row & 7;
  float zi = zbuf[0][ls][r] + zbuf[4][ls][r];
  float zf = zbuf[1][ls][r] + zbuf[5][ls][r];
  float zg = zbuf[2][ls][r] + zbuf[6][ls][r];
  float zo = zbuf[3][ls][r] + zbuf[7][ls][r];
  size_t idx = (size_t)(mb * 16 + row) * HID + nb * 16 + col;
  float cn = sigm(zf) * c[idx] + sigm(zi) * tanh1(zg);
  c[idx]    = cn;
  hout[idx] = f2bf(sigm(zo) * tanh1(cn));
}

// ---- per-timestep kernels --------------------------------------------------
// Grid: 256 blocks = dir(2) x mb(4) x nb(32); 256 threads = 8 waves per block.
__global__ __launch_bounds__(256) void lstm_step_l1(
    const long long* __restrict__ tokens, const unsigned short* __restrict__ embb,
    const unsigned short* __restrict__ W1f, const unsigned short* __restrict__ U1f,
    const float* __restrict__ b1f, const unsigned short* __restrict__ h1f_prev,
    unsigned short* __restrict__ h1f_cur, float* __restrict__ c1f,
    const unsigned short* __restrict__ W1b, const unsigned short* __restrict__ U1b,
    const float* __restrict__ b1b, const unsigned short* __restrict__ h1b_prev,
    unsigned short* __restrict__ h1b_cur, float* __restrict__ c1b, int t) {
  __shared__ float zbuf[8][32][8];            // 8 KB

  int bid = blockIdx.x;
  int dir = bid >> 7, rem = bid & 127;
  int mb = rem >> 5, nb = rem & 31;
  int wave = threadIdx.x >> 5, lane = threadIdx.x & 31;
  int g = wave & 3, s = wave >> 2;            // gate, K-half (wave-uniform)

  const unsigned short* Wp    = dir ? W1b : W1f;
  const unsigned short* Up    = dir ? U1b : U1f;
  const float*          bias  = dir ? b1b : b1f;
  const unsigned short* hprev = dir ? h1b_prev : h1f_prev;
  unsigned short*       hout  = dir ? h1b_cur : h1f_cur;
  float*                c     = dir ? c1b : c1f;

  int t_eff = dir ? (SEQT - 1 - t) : t;
  int koffA = (lane & 16) ? 8 : 0;
  int brow  = mb * 16 + (lane & 15);
  int nbg   = nb + 32 * g;

  v8f acc;
  if (s == 0) {                               // bias folded into K-half 0
    float bi = bias[g * HID + nb * 16 + (lane & 15)];
#pragma unroll
    for (int r = 0; r < 8; ++r) acc[r] = bi;
  } else {
#pragma unroll
    for (int r = 0; r < 8; ++r) acc[r] = 0.0f;
  }

  const unsigned short* hrow = hprev + (size_t)brow * HID;
  if (s == 0) {                               // x@W1 (8 chunks) + U1 chunks [0,4)
    long long tok = tokens[(size_t)brow * SEQT + t_eff];
    gemm_part(acc, embb + (size_t)tok * EMBED, Wp, EMBED / 32, 0, 8, nbg, lane, koffA);
    gemm_part(acc, hrow, Up, HID / 32, 0, 4, nbg, lane, koffA);
  } else {                                    // U1 chunks [4,16)
    gemm_part(acc, hrow, Up, HID / 32, 4, 16, nbg, lane, koffA);
  }
  tile_epilogue(zbuf, acc, wave, lane, c, hout, mb, nb, threadIdx.x);
}

__global__ __launch_bounds__(256) void lstm_step_l2(
    const unsigned short* __restrict__ h1f, const unsigned short* __restrict__ W2f,
    const unsigned short* __restrict__ U2f, const float* __restrict__ b2f,
    const unsigned short* __restrict__ h2f_prev, unsigned short* __restrict__ h2f_cur,
    float* __restrict__ c2f,
    const unsigned short* __restrict__ h1b, const unsigned short* __restrict__ W2b,
    const unsigned short* __restrict__ U2b, const float* __restrict__ b2b,
    const unsigned short* __restrict__ h2b_prev, unsigned short* __restrict__ h2b_cur,
    float* __restrict__ c2b) {
  __shared__ float zbuf[8][32][8];

  int bid = blockIdx.x;
  int dir = bid >> 7, rem = bid & 127;
  int mb = rem >> 5, nb = rem & 31;
  int wave = threadIdx.x >> 5, lane = threadIdx.x & 31;
  int g = wave & 3, s = wave >> 2;

  const unsigned short* A1    = dir ? h1b : h1f;
  const unsigned short* Wp    = dir ? W2b : W2f;
  const unsigned short* Up    = dir ? U2b : U2f;
  const float*          bias  = dir ? b2b : b2f;
  const unsigned short* hprev = dir ? h2b_prev : h2f_prev;
  unsigned short*       hout  = dir ? h2b_cur : h2f_cur;
  float*                c     = dir ? c2b : c2f;

  int koffA = (lane & 16) ? 8 : 0;
  int brow  = mb * 16 + (lane & 15);
  int nbg   = nb + 32 * g;

  v8f acc;
  if (s == 0) {
    float bi = bias[g * HID + nb * 16 + (lane & 15)];
#pragma unroll
    for (int r = 0; r < 8; ++r) acc[r] = bi;
  } else {
#pragma unroll
    for (int r = 0; r < 8; ++r) acc[r] = 0.0f;
  }

  if (s == 0) {                               // h1@W2 (16 chunks)
    gemm_part(acc, A1 + (size_t)brow * HID, Wp, HID / 32, 0, 16, nbg, lane, koffA);
  } else {                                    // h2_prev@U2 (16 chunks)
    gemm_part(acc, hprev + (size_t)brow * HID, Up, HID / 32, 0, 16, nbg, lane, koffA);
  }
  tile_epilogue(zbuf, acc, wave, lane, c, hout, mb, nb, threadIdx.x);
}

// ---- setup kernels ---------------------------------------------------------
__global__ void cvt_bf16_k(const float* __restrict__ src,
                           unsigned short* __restrict__ dst, int n) {
  int i = blockIdx.x * blockDim.x + threadIdx.x;
  if (i < n) dst[i] = f2bf(src[i]);
}

// Repack [K, 2048] fp32 -> fragment-major bf16: chunk = nbAll*(K/32)+kb holds
// one 32x16 B-fragment, element (lane l, elem i) at offset l*16+i, mapping
// k = kb*32 + (l&16 ? 8:0) + (i&7) + (i&8 ? 16:0), n = nbAll*16 + (l&15).
__global__ void repack_w_k(const float* __restrict__ src,
                           unsigned short* __restrict__ dst, int K) {
  int e = blockIdx.x * blockDim.x + threadIdx.x;
  int total = K * GATES4;
  if (e >= total) return;
  int chunk = e >> 9, within = e & 511;
  int l = within >> 4, i = within & 15;
  int kbN = K >> 5;
  int nbAll = chunk / kbN;
  int kb    = chunk - nbAll * kbN;
  int n = nbAll * 16 + (l & 15);
  int k = kb * 32 + ((l & 16) ? 8 : 0) + (i & 7) + ((i & 8) ? 16 : 0);
  dst[e] = f2bf(src[(size_t)k * GATES4 + n]);
}

__global__ void zero_u32_k(unsigned int* p, int n) {
  int i = blockIdx.x * blockDim.x + threadIdx.x;
  if (i < n) p[i] = 0u;
}

// ---- final dense head:  [64, 1024] @ [1024, 5] + bias ----------------------
__global__ void dense_out_k(const unsigned short* __restrict__ h2f,
                            const unsigned short* __restrict__ h2b,
                            const float* __restrict__ Wd,
                            const float* __restrict__ bd,
                            float* __restrict__ out) {
  int tid = threadIdx.x;                      // 320 = 64*5
  if (tid >= NB * 5) return;
  int b = tid / 5, j = tid % 5;
  float s = bd[j];
  for (int k = 0; k < HID; ++k) s += bf2f(h2f[(size_t)b * HID + k]) * Wd[k * 5 + j];
  for (int k = 0; k < HID; ++k) s += bf2f(h2b[(size_t)b * HID + k]) * Wd[(HID + k) * 5 + j];
  out[b * 5 + j] = s;
}

// ---------------------------------------------------------------------------
extern "C" void kernel_launch(void* const* d_in, const int* in_sizes, int n_in,
                              void* d_out, int out_size, void* d_ws, size_t ws_size,
                              hipStream_t stream) {
  (void)in_sizes; (void)n_in; (void)out_size; (void)ws_size;

  const long long* tokens = (const long long*)d_in[0];   // int64 per reference
  const float* emb = (const float*)d_in[1];
  const float* W1f = (const float*)d_in[2];
  const float* U1f = (const float*)d_in[3];
  const float* b1f = (const float*)d_in[4];
  const float* W2f = (const float*)d_in[5];
  const float* U2f = (const float*)d_in[6];
  const float* b2f = (const float*)d_in[7];
  const float* W1b = (const float*)d_in[8];
  const float* U1b = (const float*)d_in[9];
  const float* b1b = (const float*)d_in[10];
  const float* W2b = (const float*)d_in[11];
  const float* U2b = (const float*)d_in[12];
  const float* b2b = (const float*)d_in[13];
  const float* Wd  = (const float*)d_in[14];
  const float* bd  = (const float*)d_in[15];
  float* out = (float*)d_out;

  // ---- workspace carve-up (256B aligned) ----
  char* ws = (char*)d_ws;
  size_t off = 0;
  auto carve = [&](size_t bytes) -> char* {
    char* p = ws + off;
    off += (bytes + 255) & ~(size_t)255;
    return p;
  };
  unsigned short* embb = (unsigned short*)carve((size_t)VOCAB * EMBED * 2);
  unsigned short* W1fp = (unsigned short*)carve((size_t)EMBED * GATES4 * 2);
  unsigned short* U1fp = (unsigned short*)carve((size_t)HID * GATES4 * 2);
  unsigned short* W2fp = (unsigned short*)carve((size_t)HID * GATES4 * 2);
  unsigned short* U2fp = (unsigned short*)carve((size_t)HID * GATES4 * 2);
  unsigned short* W1bp = (unsigned short*)carve((size_t)EMBED * GATES4 * 2);
  unsigned short* U1bp = (unsigned short*)carve((size_t)HID * GATES4 * 2);
  unsigned short* W2bp = (unsigned short*)carve((size_t)HID * GATES4 * 2);
  unsigned short* U2bp = (unsigned short*)carve((size_t)HID * GATES4 * 2);
  // state region (zeroed each launch): 8 h buffers (bf16) + 4 c buffers (f32)
  char* state = ws + off;
  const size_t HB = (size_t)NB * HID * 2;   // 64KB each, 256B-multiple
  const size_t CB = (size_t)NB * HID * 4;
  unsigned short* h1fB[2] = {(unsigned short*)carve(HB), (unsigned short*)carve(HB)};
  unsigned short* h1bB[2] = {(unsigned short*)carve(HB), (unsigned short*)carve(HB)};
  unsigned short* h2fB[2] = {(unsigned short*)carve(HB), (unsigned short*)carve(HB)};
  unsigned short* h2bB[2] = {(unsigned short*)carve(HB), (unsigned short*)carve(HB)};
  float* c1f = (float*)carve(CB);
  float* c1b = (float*)carve(CB);
  float* c2f = (float*)carve(CB);
  float* c2b = (float*)carve(CB);
  size_t state_u32 = (8 * HB + 4 * CB) / 4;

  // ---- setup: convert embedding, repack weights, zero state ----
  {
    int n = VOCAB * EMBED;
    cvt_bf16_k<<<(n + 255) / 256, 256, 0, stream>>>(emb, embb, n);
  }
  repack_w_k<<<(EMBED * GATES4 + 255) / 256, 256, 0, stream>>>(W1f, W1fp, EMBED);
  repack_w_k<<<(HID   * GATES4 + 255) / 256, 256, 0, stream>>>(U1f, U1fp, HID);
  repack_w_k<<<(HID   * GATES4 + 255) / 256, 256, 0, stream>>>(W2f, W2fp, HID);
  repack_w_k<<<(HID   * GATES4 + 255) / 256, 256, 0, stream>>>(U2f, U2fp, HID);
  repack_w_k<<<(EMBED * GATES4 + 255) / 256, 256, 0, stream>>>(W1b, W1bp, EMBED);
  repack_w_k<<<(HID   * GATES4 + 255) / 256, 256, 0, stream>>>(U1b, U1bp, HID);
  repack_w_k<<<(HID   * GATES4 + 255) / 256, 256, 0, stream>>>(W2b, W2bp, HID);
  repack_w_k<<<(HID   * GATES4 + 255) / 256, 256, 0, stream>>>(U2b, U2bp, HID);
  zero_u32_k<<<((int)state_u32 + 255) / 256, 256, 0, stream>>>((unsigned int*)state,
                                                               (int)state_u32);

  // ---- recurrence: 512 steps, double-buffered h, L1 then L2 ----
  for (int t = 0; t < SEQT; ++t) {
    int p = t & 1, q = p ^ 1;
    lstm_step_l1<<<256, 256, 0, stream>>>(
        tokens, embb,
        W1fp, U1fp, b1f, h1fB[p], h1fB[q], c1f,
        W1bp, U1bp, b1b, h1bB[p], h1bB[q], c1b, t);
    lstm_step_l2<<<256, 256, 0, stream>>>(
        h1fB[q], W2fp, U2fp, b2f, h2fB[p], h2fB[q], c2f,
        h1bB[q], W2bp, U2bp, b2b, h2bB[p], h2bB[q], c2b);
  }

  // final h2 landed in buffer index 0 (t=511 odd: cur = 0)
  dense_out_k<<<1, 320, 0, stream>>>(h2fB[0], h2bB[0], Wd, bd, out);
}
